// DomainExpert_87488483820293
// MI455X (gfx1250) — matrix operand order repaired and checked by reading, not compiled, not run on hardware
//
#include <hip/hip_runtime.h>
#include <hip/hip_bf16.h>
#include <math.h>

typedef __bf16 bf16;
typedef __attribute__((ext_vector_type(16))) __bf16 bf16x16;
typedef __attribute__((ext_vector_type(8)))  __bf16 bf16x8;
typedef __attribute__((ext_vector_type(8)))  float  f32x8;

union FragAB { bf16x16 v; bf16x8 h[2]; };

constexpr int TOK = 16384;   // 4 * 4096 tokens
constexpr int SEQ = 4096;

// Tensor Data Mover availability (this toolchain: clang-23 6-arg builtin)
#if defined(__has_builtin)
#  if __has_builtin(__builtin_amdgcn_tensor_load_to_lds) && __has_builtin(__builtin_amdgcn_s_wait_tensorcnt)
#    define USE_TDM 1
#  endif
#endif
#ifndef USE_TDM
#  define USE_TDM 0
#endif

#if USE_TDM
typedef __attribute__((ext_vector_type(4))) unsigned int u32x4;
typedef __attribute__((ext_vector_type(8))) int          i32x8;
typedef __attribute__((ext_vector_type(4))) int          i32x4;

// Generic LDS pointer: its low 32 bits are the LDS byte offset.
__device__ __forceinline__ unsigned lds_off(const void* p) {
  return (unsigned)(uintptr_t)p;
}

// DMA a 2D tile into LDS: 128 rows x 32 bf16 (64B), global row stride = K
// elements, with HW padding of 4 DWORDs (16B) after every 16 DWORDs (64B)
// -> LDS row stride 80B, matching bf16[128][40].
__device__ __forceinline__ void tdm_load_tile(unsigned ldsAddr, const bf16* gptr, int K) {
  const unsigned long long ga = (unsigned long long)(uintptr_t)gptr;
  u32x4 g0;
  g0[0] = 0x1u;                                   // count=1, user descriptor
  g0[1] = ldsAddr;                                // lds_addr (bytes)
  g0[2] = (unsigned)(ga & 0xffffffffu);           // global_addr[31:0]
  g0[3] = (unsigned)((ga >> 32) & 0x01ffffffu)    // global_addr[56:32]
          | (2u << 30);                           // type = 2 ("image")
  i32x8 g1;
  g1[0] = (1 << 16)                               // data_size = 2B
        | (1 << 20)                               // pad_enable
        | (3 << 22)                               // pad_interval: 16 DWORDs
        | (3 << 25);                              // pad_amount: 4 DWORDs
  g1[1] = (K & 0xffff) << 16;                     // tensor_dim0[15:0]
  g1[2] = ((unsigned)K >> 16) | (128u << 16);     // tensor_dim0[31:16] | tensor_dim1[15:0]=128
  g1[3] = (32 << 16);                             // tensor_dim1 hi=0 | tile_dim0 = 32
  g1[4] = 128;                                    // tile_dim1 = 128 rows (tile_dim2 = 0)
  g1[5] = K;                                      // tensor_dim0_stride[31:0]
  g1[6] = 0;                                      // stride0 hi | stride1 lo
  g1[7] = 0;
  const i32x4 z4 = {0, 0, 0, 0};                  // 2D tensor: groups 2/3 unused
  const i32x8 z8 = {0, 0, 0, 0, 0, 0, 0, 0};      // extra group (clang-23 form)
  __builtin_amdgcn_tensor_load_to_lds(g0, g1, z4, z4, z8, 0);
}
#endif

// ---------------------------------------------------------------------------
// Weight convert + transpose: in fp32 [K][N] -> out bf16 [N][K]
// ---------------------------------------------------------------------------
__global__ void wt_transpose_kernel(const float* __restrict__ in, bf16* __restrict__ out,
                                    int K, int N) {
  int idx = blockIdx.x * 256 + threadIdx.x;
  if (idx >= K * N) return;
  int n = idx / K, k = idx - n * K;
  out[idx] = (bf16)in[(size_t)k * N + n];
}

// ---------------------------------------------------------------------------
// LayerNorm over 1024 columns, fp32 in -> bf16 out. One block per row.
// ---------------------------------------------------------------------------
__global__ __launch_bounds__(256)
void layernorm_bf16_kernel(const float* __restrict__ in, const float* __restrict__ g,
                           const float* __restrict__ be, bf16* __restrict__ out) {
  __shared__ float s1[256], s2[256];
  const int row = blockIdx.x, tid = threadIdx.x;
  const float4 v4 = *(const float4*)(in + (size_t)row * 1024 + tid * 4);
  s1[tid] = v4.x + v4.y + v4.z + v4.w;
  s2[tid] = v4.x * v4.x + v4.y * v4.y + v4.z * v4.z + v4.w * v4.w;
  __syncthreads();
  for (int s = 128; s > 0; s >>= 1) {
    if (tid < s) { s1[tid] += s1[tid + s]; s2[tid] += s2[tid + s]; }
    __syncthreads();
  }
  const float mu  = s1[0] * (1.0f / 1024.0f);
  const float var = s2[0] * (1.0f / 1024.0f) - mu * mu;
  const float inv = rsqrtf(var + 1e-5f);
  const int c = tid * 4;
  bf16* o = out + (size_t)row * 1024 + c;
  o[0] = (bf16)((v4.x - mu) * inv * g[c + 0] + be[c + 0]);
  o[1] = (bf16)((v4.y - mu) * inv * g[c + 1] + be[c + 1]);
  o[2] = (bf16)((v4.z - mu) * inv * g[c + 2] + be[c + 2]);
  o[3] = (bf16)((v4.w - mu) * inv * g[c + 3] + be[c + 3]);
}

// ---------------------------------------------------------------------------
// Generic bf16 WMMA GEMM: C[M,N] = A[M,K] * BT[N,K]^T + bias (+epilogue)
// EPI: 0 = store bf16, 1 = GELU then store bf16, 2 = + res then store fp32
// Block: 256 thr (8 waves), tile 128x128, K-step 32, double-buffered LDS.
// Tile staging: Tensor Data Mover (one wave issues, TENSORcnt + barrier),
// fallback: reg-staged vector loads.
// ---------------------------------------------------------------------------
template<int EPI>
__global__ __launch_bounds__(256)
void gemm_bf16_kernel(const bf16* __restrict__ A, const bf16* __restrict__ BT,
                      const float* __restrict__ bias, const float* __restrict__ res,
                      bf16* __restrict__ outB, float* __restrict__ outF,
                      int M, int N, int K) {
  __shared__ bf16 As[2][128][40];
  __shared__ bf16 Bs[2][128][40];
  const int tid = threadIdx.x, lane = tid & 31, wave = tid >> 5;
  const int lo = lane & 15, hi = lane >> 4;
  const int wm = wave >> 1, wn = wave & 1;
  const int m0 = blockIdx.y * 128, n0 = blockIdx.x * 128;

  const f32x8 fz = {0.f, 0.f, 0.f, 0.f, 0.f, 0.f, 0.f, 0.f};
  f32x8 acc[2][4];
  for (int i = 0; i < 2; ++i)
    for (int j = 0; j < 4; ++j) acc[i][j] = fz;

#if USE_TDM
  const bf16* gA = A  + (size_t)m0 * K;
  const bf16* gB = BT + (size_t)n0 * K;
  const unsigned ldsA[2] = { lds_off(&As[0][0][0]), lds_off(&As[1][0][0]) };
  const unsigned ldsB[2] = { lds_off(&Bs[0][0][0]), lds_off(&Bs[1][0][0]) };
  if (wave == 0) {                          // prime buffer 0 via TDM
    tdm_load_tile(ldsA[0], gA, K);
    tdm_load_tile(ldsB[0], gB, K);
    __builtin_amdgcn_s_wait_tensorcnt((short)0);
  }
  __syncthreads();
#else
  const int lr = tid >> 1, lc = (tid & 1) * 16;
  const bf16* ga = A  + (size_t)(m0 + lr) * K + lc;
  const bf16* gb = BT + (size_t)(n0 + lr) * K + lc;
  {                                         // prime buffer 0 via regs
    bf16x8 ra0 = *(const bf16x8*)(ga);
    bf16x8 ra1 = *(const bf16x8*)(ga + 8);
    bf16x8 rb0 = *(const bf16x8*)(gb);
    bf16x8 rb1 = *(const bf16x8*)(gb + 8);
    *(bf16x8*)&As[0][lr][lc]     = ra0;
    *(bf16x8*)&As[0][lr][lc + 8] = ra1;
    *(bf16x8*)&Bs[0][lr][lc]     = rb0;
    *(bf16x8*)&Bs[0][lr][lc + 8] = rb1;
  }
  __syncthreads();
#endif

  int buf = 0;
  for (int kb = 0; kb < K; kb += 32) {
    const bool more = (kb + 32) < K;
#if USE_TDM
    if (more && wave == 0) {                // start DMA of next tile early
      tdm_load_tile(ldsA[buf ^ 1], gA + kb + 32, K);
      tdm_load_tile(ldsB[buf ^ 1], gB + kb + 32, K);
    }
#else
    bf16x8 ra0, ra1, rb0, rb1;
    if (more) {                             // start loads of next tile early
      ra0 = *(const bf16x8*)(ga + kb + 32);
      ra1 = *(const bf16x8*)(ga + kb + 40);
      rb0 = *(const bf16x8*)(gb + kb + 32);
      rb1 = *(const bf16x8*)(gb + kb + 40);
    }
#endif

    FragAB af[2], bf_[4];
#pragma unroll
    for (int i = 0; i < 2; ++i) {
      const int rm = wm * 32 + i * 16 + lo;
      af[i].h[0] = *(const bf16x8*)&As[buf][rm][hi * 8];        // K = hi*8 + 0..7
      af[i].h[1] = *(const bf16x8*)&As[buf][rm][16 + hi * 8];   // K = 16 + hi*8 + 0..7
    }
#pragma unroll
    for (int j = 0; j < 4; ++j) {
      const int rn = wn * 64 + j * 16 + lo;
      bf_[j].h[0] = *(const bf16x8*)&Bs[buf][rn][hi * 8];       // K = hi*8 + 0..7
      bf_[j].h[1] = *(const bf16x8*)&Bs[buf][rn][16 + hi * 8];  // K = 16 + hi*8 + 0..7
    }
#pragma unroll
    for (int i = 0; i < 2; ++i)
#pragma unroll
      for (int j = 0; j < 4; ++j)
        acc[i][j] = __builtin_amdgcn_wmma_f32_16x16x32_bf16(
            false, af[i].v, false, bf_[j].v, (short)0, acc[i][j], false, false);

#if USE_TDM
    if (more && wave == 0) __builtin_amdgcn_s_wait_tensorcnt((short)0);
#else
    if (more) {                             // stash next tile in other buffer
      *(bf16x8*)&As[buf ^ 1][lr][lc]     = ra0;
      *(bf16x8*)&As[buf ^ 1][lr][lc + 8] = ra1;
      *(bf16x8*)&Bs[buf ^ 1][lr][lc]     = rb0;
      *(bf16x8*)&Bs[buf ^ 1][lr][lc + 8] = rb1;
    }
#endif
    __syncthreads();
    buf ^= 1;
  }

  for (int j = 0; j < 4; ++j) {
    const int gcol = n0 + wn * 64 + j * 16 + lo;
    const float bv = bias[gcol];
    for (int i = 0; i < 2; ++i) {
#pragma unroll
      for (int rr = 0; rr < 8; ++rr) {
        const int grow = m0 + wm * 32 + i * 16 + hi * 8 + rr;
        const size_t idx = (size_t)grow * N + gcol;
        float v = acc[i][j][rr] + bv;
        if (EPI == 1) v = 0.5f * v * (1.0f + erff(v * 0.70710678118654752f));
        if (EPI == 2) outF[idx] = v + res[idx];
        else          outB[idx] = (bf16)v;
      }
    }
  }
}

// ---------------------------------------------------------------------------
// k softmax stats over sequence axis: per (b, channel) max & sum(exp).
// ---------------------------------------------------------------------------
__global__ __launch_bounds__(256)
void kstats_kernel(const bf16* __restrict__ qkv, float* __restrict__ kmax,
                   float* __restrict__ ksum) {
  __shared__ float sh[256];
  const int bc = blockIdx.x, tid = threadIdx.x;
  const int b = bc >> 10, c = bc & 1023;
  const bf16* base = qkv + (size_t)b * SEQ * 3072 + 1024 + c;
  float vals[16];
  float lm = -3.0e38f;
#pragma unroll
  for (int i = 0; i < 16; ++i) {
    vals[i] = (float)base[(size_t)(i * 256 + tid) * 3072];
    lm = fmaxf(lm, vals[i]);
  }
  sh[tid] = lm; __syncthreads();
  for (int s = 128; s > 0; s >>= 1) { if (tid < s) sh[tid] = fmaxf(sh[tid], sh[tid + s]); __syncthreads(); }
  const float mx = sh[0]; __syncthreads();
  float ls = 0.f;
#pragma unroll
  for (int i = 0; i < 16; ++i) ls += expf(vals[i] - mx);
  sh[tid] = ls; __syncthreads();
  for (int s = 128; s > 0; s >>= 1) { if (tid < s) sh[tid] += sh[tid + s]; __syncthreads(); }
  if (tid == 0) { kmax[bc] = mx; ksum[bc] = sh[0]; }
}

// ---------------------------------------------------------------------------
// In-place: q softmax per (token, head) over 64 dims; k normalize by
// sequence-axis stats. v untouched. One block per token.
// ---------------------------------------------------------------------------
__global__ __launch_bounds__(256)
void softmax_apply_kernel(bf16* __restrict__ qkv, const float* __restrict__ kmax,
                          const float* __restrict__ ksum) {
  __shared__ float sh[256];
  __shared__ float hstat[16];
  const int tok = blockIdx.x, tid = threadIdx.x;
  const int b = tok >> 12;
  bf16* row = qkv + (size_t)tok * 3072;
  const int head = tid >> 4, li = tid & 15;
  const int qc = head * 64 + li * 4;
  float qv[4];
#pragma unroll
  for (int u = 0; u < 4; ++u) qv[u] = (float)row[qc + u];
  float lm = fmaxf(fmaxf(qv[0], qv[1]), fmaxf(qv[2], qv[3]));
  sh[tid] = lm; __syncthreads();
  if (li == 0) {
    float m = sh[tid];
    for (int t = 1; t < 16; ++t) m = fmaxf(m, sh[tid + t]);
    hstat[head] = m;
  }
  __syncthreads();
  const float hm = hstat[head];
  float ls = 0.f;
#pragma unroll
  for (int u = 0; u < 4; ++u) { qv[u] = expf(qv[u] - hm); ls += qv[u]; }
  sh[tid] = ls; __syncthreads();
  if (li == 0) {
    float s2 = 0.f;
    for (int t = 0; t < 16; ++t) s2 += sh[tid + t];
    hstat[head] = s2;
  }
  __syncthreads();
  const float inv = 1.0f / hstat[head];
#pragma unroll
  for (int u = 0; u < 4; ++u) row[qc + u] = (bf16)(qv[u] * inv);
#pragma unroll
  for (int u = 0; u < 4; ++u) {
    const int c = tid * 4 + u;
    const float kv = (float)row[1024 + c];
    row[1024 + c] = (bf16)(expf(kv - kmax[(b << 10) + c]) / ksum[(b << 10) + c]);
  }
}

// ---------------------------------------------------------------------------
// kv accumulator init
// ---------------------------------------------------------------------------
__global__ void kv_init_kernel(float* __restrict__ kvacc) {
  kvacc[blockIdx.x * 256 + threadIdx.x] = 0.0f;
}

// ---------------------------------------------------------------------------
// kv = k'^T v per (b,h): 64x64 over K=S. Grid (64 bh, 8 s-chunks of 512).
// ---------------------------------------------------------------------------
__global__ __launch_bounds__(256)
void kv_partial_kernel(const bf16* __restrict__ qkv, float* __restrict__ kvacc) {
  __shared__ bf16 kT[64][40];   // [d][s]
  __shared__ bf16 vT[64][40];   // [e][s]
  const int tid = threadIdx.x, lane = tid & 31, wave = tid >> 5;
  const int lo = lane & 15, hi = lane >> 4;
  const int bh = blockIdx.x, b = bh >> 4, h = bh & 15;
  const int sbase = blockIdx.y * 512;
  const int di = wave >> 2, ej = wave & 3;
  const int sIdx = tid >> 3, dg = tid & 7;

  const f32x8 fz = {0.f, 0.f, 0.f, 0.f, 0.f, 0.f, 0.f, 0.f};
  f32x8 acc[2]; acc[0] = fz; acc[1] = fz;

  for (int ks = 0; ks < 512; ks += 32) {
    const size_t token = (size_t)b * SEQ + sbase + ks + sIdx;
    const bf16x8 kk = *(const bf16x8*)(qkv + token * 3072 + 1024 + h * 64 + dg * 8);
    const bf16x8 vv = *(const bf16x8*)(qkv + token * 3072 + 2048 + h * 64 + dg * 8);
#pragma unroll
    for (int u = 0; u < 8; ++u) {  // transpose into LDS
      kT[dg * 8 + u][sIdx] = kk[u];
      vT[dg * 8 + u][sIdx] = vv[u];
    }
    __syncthreads();
    FragAB a[2], bv;
#pragma unroll
    for (int i = 0; i < 2; ++i) {
      const int d = di * 32 + i * 16 + lo;
      a[i].h[0] = *(const bf16x8*)&kT[d][hi * 8];
      a[i].h[1] = *(const bf16x8*)&kT[d][16 + hi * 8];
    }
    const int e = ej * 16 + lo;
    bv.h[0] = *(const bf16x8*)&vT[e][hi * 8];
    bv.h[1] = *(const bf16x8*)&vT[e][16 + hi * 8];
#pragma unroll
    for (int i = 0; i < 2; ++i)
      acc[i] = __builtin_amdgcn_wmma_f32_16x16x32_bf16(
          false, a[i].v, false, bv.v, (short)0, acc[i], false, false);
    __syncthreads();
  }
#pragma unroll
  for (int i = 0; i < 2; ++i)
#pragma unroll
    for (int rr = 0; rr < 8; ++rr) {
      const int d = di * 32 + i * 16 + hi * 8 + rr;
      const int e = ej * 16 + lo;
      atomicAdd(&kvacc[((size_t)bh * 64 + d) * 64 + e], acc[i][rr]);
    }
}

// ---------------------------------------------------------------------------
// kvT[bh][e][d] = bf16(kvacc[bh][d][e])
// ---------------------------------------------------------------------------
__global__ void kv_finalize_kernel(const float* __restrict__ kvacc, bf16* __restrict__ kvT) {
  const int idx = blockIdx.x * 256 + threadIdx.x;  // 64*64*64 total
  const int bh = idx >> 12, rem = idx & 4095;
  const int d = rem >> 6, e = rem & 63;
  kvT[((size_t)bh << 12) + e * 64 + d] = (bf16)kvacc[idx];
}

// ---------------------------------------------------------------------------
// attn[s, h*64+e] = sum_d q'[s, h*64+d] * kv[d][e]   (K = 64, per (b,h))
// ---------------------------------------------------------------------------
__global__ __launch_bounds__(256)
void attn_ctx_kernel(const bf16* __restrict__ qkv, const bf16* __restrict__ kvT,
                     bf16* __restrict__ attn) {
  const int tid = threadIdx.x, lane = tid & 31, wave = tid >> 5;
  const int lo = lane & 15, hi = lane >> 4;
  const int bh = blockIdx.x, b = bh >> 4, h = bh & 15;
  const int s0 = blockIdx.y * 128 + wave * 16;

  const f32x8 fz = {0.f, 0.f, 0.f, 0.f, 0.f, 0.f, 0.f, 0.f};
  f32x8 acc[4]; for (int j = 0; j < 4; ++j) acc[j] = fz;

#pragma unroll
  for (int k0 = 0; k0 < 64; k0 += 32) {
    FragAB a;
    const size_t arow = ((size_t)(b * SEQ + s0 + lo)) * 3072 + h * 64 + k0;
    a.h[0] = *(const bf16x8*)(qkv + arow + hi * 8);
    a.h[1] = *(const bf16x8*)(qkv + arow + 16 + hi * 8);
    FragAB bf_[4];
#pragma unroll
    for (int j = 0; j < 4; ++j) {
      const size_t brow = ((size_t)bh * 64 + j * 16 + lo) * 64 + k0;
      bf_[j].h[0] = *(const bf16x8*)(kvT + brow + hi * 8);
      bf_[j].h[1] = *(const bf16x8*)(kvT + brow + 16 + hi * 8);
    }
#pragma unroll
    for (int j = 0; j < 4; ++j)
      acc[j] = __builtin_amdgcn_wmma_f32_16x16x32_bf16(
          false, a.v, false, bf_[j].v, (short)0, acc[j], false, false);
  }
#pragma unroll
  for (int j = 0; j < 4; ++j)
#pragma unroll
    for (int rr = 0; rr < 8; ++rr) {
      const int row = b * SEQ + s0 + hi * 8 + rr;
      const int col = h * 64 + j * 16 + lo;
      attn[(size_t)row * 1024 + col] = (bf16)acc[j][rr];
    }
}

// ---------------------------------------------------------------------------
// Host orchestration
// ---------------------------------------------------------------------------
extern "C" void kernel_launch(void* const* d_in, const int* in_sizes, int n_in,
                              void* d_out, int out_size, void* d_ws, size_t ws_size,
                              hipStream_t stream) {
  const float* x    = (const float*)d_in[0];
  const float* wqkv = (const float*)d_in[1];
  const float* bqkv = (const float*)d_in[2];
  const float* wao  = (const float*)d_in[3];
  const float* bao  = (const float*)d_in[4];
  const float* wf1  = (const float*)d_in[5];
  const float* bf1  = (const float*)d_in[6];
  const float* wf2  = (const float*)d_in[7];
  const float* bf2  = (const float*)d_in[8];
  const float* g1   = (const float*)d_in[9];
  const float* be1  = (const float*)d_in[10];
  const float* g2   = (const float*)d_in[11];
  const float* be2  = (const float*)d_in[12];

  char* ws = (char*)d_ws;
  size_t off = 0;
  auto alloc = [&](size_t bytes) -> char* {
    char* p = ws + off;
    off = (off + bytes + 255) & ~(size_t)255;
    return p;
  };
  bf16*  wTqkv = (bf16*)alloc(3072ull * 1024 * 2);
  bf16*  wTao  = (bf16*)alloc(1024ull * 1024 * 2);
  bf16*  wTf1  = (bf16*)alloc(4096ull * 1024 * 2);
  bf16*  wTf2  = (bf16*)alloc(1024ull * 4096 * 2);
  bf16*  xn    = (bf16*)alloc((size_t)TOK * 1024 * 2);   // dies after qkv GEMM
  bf16*  qkv   = (bf16*)alloc((size_t)TOK * 3072 * 2);   // dies after attn_ctx
  float* kmax  = (float*)alloc(4096 * 4);
  float* ksum  = (float*)alloc(4096 * 4);
  float* kvacc = (float*)alloc(64ull * 64 * 64 * 4);
  bf16*  kvT   = (bf16*)alloc(64ull * 64 * 64 * 2);
  bf16*  attn  = (bf16*)alloc((size_t)TOK * 1024 * 2);
  float* y     = (float*)alloc((size_t)TOK * 1024 * 4);
  bf16*  z     = (bf16*)alloc((size_t)TOK * 1024 * 2);
  bf16*  ff1   = xn;  // xn(32MB)+qkv(96MB) contiguous region == exactly TOK*4096 bf16

  // weights -> bf16, transposed to [N][K]
  wt_transpose_kernel<<<(1024 * 3072 + 255) / 256, 256, 0, stream>>>(wqkv, wTqkv, 1024, 3072);
  wt_transpose_kernel<<<(1024 * 1024 + 255) / 256, 256, 0, stream>>>(wao, wTao, 1024, 1024);
  wt_transpose_kernel<<<(1024 * 4096 + 255) / 256, 256, 0, stream>>>(wf1, wTf1, 1024, 4096);
  wt_transpose_kernel<<<(4096 * 1024 + 255) / 256, 256, 0, stream>>>(wf2, wTf2, 4096, 1024);

  // LN1 -> xn ; qkv = xn @ Wqkv + b
  layernorm_bf16_kernel<<<TOK, 256, 0, stream>>>(x, g1, be1, xn);
  gemm_bf16_kernel<0><<<dim3(24, 128), 256, 0, stream>>>(xn, wTqkv, bqkv, nullptr,
                                                         qkv, nullptr, TOK, 3072, 1024);
  // softmaxes
  kstats_kernel<<<4096, 256, 0, stream>>>(qkv, kmax, ksum);
  softmax_apply_kernel<<<TOK, 256, 0, stream>>>(qkv, kmax, ksum);

  // kv einsum + context
  kv_init_kernel<<<1024, 256, 0, stream>>>(kvacc);
  kv_partial_kernel<<<dim3(64, 8), 256, 0, stream>>>(qkv, kvacc);
  kv_finalize_kernel<<<1024, 256, 0, stream>>>(kvacc, kvT);
  attn_ctx_kernel<<<dim3(64, 32), 256, 0, stream>>>(qkv, kvT, attn);

  // out projection + residual -> y ; LN2 -> z
  gemm_bf16_kernel<2><<<dim3(8, 128), 256, 0, stream>>>(attn, wTao, bao, x,
                                                        nullptr, y, TOK, 1024, 1024);
  layernorm_bf16_kernel<<<TOK, 256, 0, stream>>>(y, g2, be2, z);

  // FFN1 (+GELU) -> ff1 ; FFN2 + x -> d_out
  gemm_bf16_kernel<1><<<dim3(32, 128), 256, 0, stream>>>(z, wTf1, bf1, nullptr,
                                                         ff1, nullptr, TOK, 4096, 1024);
  gemm_bf16_kernel<2><<<dim3(8, 128), 256, 0, stream>>>(ff1, wTf2, bf2, x,
                                                        nullptr, (float*)d_out, TOK, 1024, 4096);
  (void)in_sizes; (void)n_in; (void)out_size; (void)ws_size;
}
